// VGGTFusionNormalizedGraspNet_16733192585536
// MI455X (gfx1250) — compile-verified, matrix operand order checked by reading, not run on hardware
//
#include <hip/hip_runtime.h>
#include <stdint.h>

// ---------------- problem dims (fixed by reference setup_inputs) ----------------
#define BB 4
#define NPC 20000
#define MM 1024
#define KK 16384
#define DD 768

typedef __attribute__((ext_vector_type(16))) __bf16 v16bf;
typedef __attribute__((ext_vector_type(8)))  float  v8f;

union Frag16 { v16bf bf; uint4 q[2]; };
union Acc8   { v8f v; float f[8]; };

// Load a 16x32 bf16 A/B fragment from row-major [row][K] storage (ld elements).
// 16-bit fragment layout: lane%16 -> row, lane/16 selects K {8..15,24..31} half;
// per lane: 16 bf16 = two 16-byte loads at k0+hi*8 and k0+16+hi*8.
__device__ __forceinline__ v16bf load_frag_bf16(const __bf16* __restrict__ base,
                                                int ld, int k0, int lane) {
  const __bf16* p = base + (size_t)(lane & 15) * ld + (k0 + ((lane >> 4) << 3));
  Frag16 f;
  f.q[0] = *(const uint4*)(p);
  f.q[1] = *(const uint4*)(p + 16);
  return f.bf;
}

__device__ __forceinline__ v8f wmma_bf16(v16bf a, v16bf b, v8f c) {
  // (neg_a, A, neg_b, B, c_mod, C, reuse_a, reuse_b)
  return __builtin_amdgcn_wmma_f32_16x16x32_bf16(false, a, false, b, (short)0, c,
                                                 false, false);
}

// ---------------- kernel 1: per-batch centroid + max-radius scale ----------------
__global__ __launch_bounds__(256)
void centroid_kernel(const float* __restrict__ pc, float* __restrict__ cs) {
  const int b = blockIdx.x, tid = threadIdx.x;
  __shared__ float rx[256], ry[256], rz[256];
  __shared__ float csh[3];
  const float* p = pc + (size_t)b * NPC * 3;
  float sx = 0.f, sy = 0.f, sz = 0.f;
  for (int i = tid; i < NPC; i += 256) {
    sx += p[i * 3 + 0]; sy += p[i * 3 + 1]; sz += p[i * 3 + 2];
  }
  rx[tid] = sx; ry[tid] = sy; rz[tid] = sz;
  __syncthreads();
  for (int s = 128; s > 0; s >>= 1) {
    if (tid < s) { rx[tid] += rx[tid + s]; ry[tid] += ry[tid + s]; rz[tid] += rz[tid + s]; }
    __syncthreads();
  }
  if (tid == 0) {
    csh[0] = rx[0] / (float)NPC; csh[1] = ry[0] / (float)NPC; csh[2] = rz[0] / (float)NPC;
  }
  __syncthreads();
  const float cx = csh[0], cy = csh[1], cz = csh[2];
  float mr = 0.f;
  for (int i = tid; i < NPC; i += 256) {
    float dx = p[i * 3 + 0] - cx, dy = p[i * 3 + 1] - cy, dz = p[i * 3 + 2] - cz;
    mr = fmaxf(mr, dx * dx + dy * dy + dz * dz);
  }
  __syncthreads();
  rx[tid] = mr;
  __syncthreads();
  for (int s = 128; s > 0; s >>= 1) {
    if (tid < s) rx[tid] = fmaxf(rx[tid], rx[tid + s]);
    __syncthreads();
  }
  if (tid == 0) {
    cs[b * 4 + 0] = cx; cs[b * 4 + 1] = cy; cs[b * 4 + 2] = cz;
    cs[b * 4 + 3] = fmaxf(sqrtf(rx[0]), 1e-6f);
  }
}

// ---------------- kernel 2: normalize pts and seeds ----------------
__global__ void normalize_kernel(const float* __restrict__ pts,
                                 const float* __restrict__ seeds,
                                 const float* __restrict__ cs,
                                 float* __restrict__ pts_n,
                                 float* __restrict__ seed_n) {
  const int per = KK + MM;
  int i = blockIdx.x * blockDim.x + threadIdx.x;
  if (i >= BB * per) return;
  const int b = i / per, r = i % per;
  const float cx = cs[b * 4 + 0], cy = cs[b * 4 + 1], cz = cs[b * 4 + 2];
  const float inv = 1.f / cs[b * 4 + 3];
  const float* s; float* d;
  if (r < KK) {
    s = pts + ((size_t)b * KK + r) * 3;  d = pts_n + ((size_t)b * KK + r) * 3;
  } else {
    const int rr = r - KK;
    s = seeds + ((size_t)b * MM + rr) * 3; d = seed_n + ((size_t)b * MM + rr) * 3;
  }
  d[0] = (s[0] - cx) * inv; d[1] = (s[1] - cy) * inv; d[2] = (s[2] - cz) * inv;
}

// ---------------- kernel 3: NN argmin (1 wave per seed, pts tiled in LDS) -------
#define NTILE 2048
__global__ __launch_bounds__(256)
void nn_kernel(const float* __restrict__ seed_n, const float* __restrict__ pts_n,
               int* __restrict__ idxbuf) {
  __shared__ float ptsS[NTILE * 3];
  const int tid = threadIdx.x, lane = tid & 31, w = tid >> 5;
  const int seed0 = blockIdx.x * 8;          // 8 waves -> 8 seeds, same batch
  const int b = seed0 / MM;
  const int m = (seed0 % MM) + w;
  const float sx = seed_n[(size_t)(b * MM + m) * 3 + 0];
  const float sy = seed_n[(size_t)(b * MM + m) * 3 + 1];
  const float sz = seed_n[(size_t)(b * MM + m) * 3 + 2];
  float bv = 3.4e38f; int bi = 0;
  for (int t0 = 0; t0 < KK; t0 += NTILE) {
    __syncthreads();
    const float* src = pts_n + ((size_t)b * KK + t0) * 3;
    for (int i = tid; i < NTILE * 3; i += 256) ptsS[i] = src[i];
    __syncthreads();
    for (int j = lane; j < NTILE; j += 32) {
      const float dx = sx - ptsS[j * 3 + 0];
      const float dy = sy - ptsS[j * 3 + 1];
      const float dz = sz - ptsS[j * 3 + 2];
      const float d2 = dx * dx + dy * dy + dz * dz;
      const int gi = t0 + j;
      if (d2 < bv || (d2 == bv && gi < bi)) { bv = d2; bi = gi; }
    }
  }
#pragma unroll
  for (int off = 16; off > 0; off >>= 1) {
    const float ov = __shfl_xor(bv, off, 32);
    const int   oi = __shfl_xor(bi, off, 32);
    if (ov < bv || (ov == bv && oi < bi)) { bv = ov; bi = oi; }
  }
  if (lane == 0) idxbuf[b * MM + m] = bi;
}

// ---------------- kernel 4: f32 -> bf16 weight conversion ----------------
__global__ void cvt_bf16_kernel(const float* __restrict__ src, __bf16* __restrict__ dst,
                                int n) {
  int i = blockIdx.x * blockDim.x + threadIdx.x;
  if (i < n) dst[i] = (__bf16)src[i];
}

// ---------------- kernel 5: fused gather + LN + MLP1 + concat + MLP2 -----------
// One block per 16-seed tile; 8 waves; all GEMMs via v_wmma_f32_16x16x32_bf16.
__global__ __launch_bounds__(256)
void fused_mlp_kernel(const float* __restrict__ feat, const int* __restrict__ idxbuf,
                      const float* __restrict__ gamma, const float* __restrict__ beta,
                      const __bf16* __restrict__ w1, const float* __restrict__ b1,
                      const __bf16* __restrict__ w2, const float* __restrict__ b2,
                      const __bf16* __restrict__ w3, const float* __restrict__ b3,
                      const __bf16* __restrict__ w4, const float* __restrict__ b4,
                      const float* __restrict__ seed_feat, float* __restrict__ out) {
  constexpr int XLN_LD = DD + 8;    // 776 elems -> 1552B row (16B aligned)
  constexpr int H_LD   = 256 + 8;   // 528B row
  constexpr int F_LD   = 512 + 8;   // 1040B row
  __shared__ __align__(16) __bf16 xln[16][XLN_LD];
  __shared__ __align__(16) __bf16 hbuf[16][H_LD];
  __shared__ __align__(16) __bf16 fbuf[16][F_LD];

  const int b   = blockIdx.x >> 6;          // 64 tiles per batch
  const int m0  = (blockIdx.x & 63) << 4;
  const int tid = threadIdx.x;
  const int lane = tid & 31;
  const int w = tid >> 5;
  const int n = lane & 15, hi = lane >> 4;

  // ---- Stage 1: gather nearest feat column + LayerNorm(768) -> xln (bf16) ----
  for (int ss = 0; ss < 2; ++ss) {
    const int mloc = w * 2 + ss;
    const int kidx = idxbuf[b * MM + m0 + mloc];
    float v[DD / 32];
    float s = 0.f, s2 = 0.f;
#pragma unroll
    for (int i = 0; i < DD / 32; ++i) {
      const int c = lane + 32 * i;
      const float x = feat[((size_t)b * DD + c) * KK + kidx];
      v[i] = x; s += x; s2 += x * x;
    }
#pragma unroll
    for (int off = 16; off > 0; off >>= 1) {
      s  += __shfl_xor(s, off, 32);
      s2 += __shfl_xor(s2, off, 32);
    }
    const float mu   = s / (float)DD;
    const float var  = s2 / (float)DD - mu * mu;
    const float rstd = rsqrtf(var + 1e-5f);
#pragma unroll
    for (int i = 0; i < DD / 32; ++i) {
      const int c = lane + 32 * i;
      xln[mloc][c] = (__bf16)(((v[i] - mu) * rstd) * gamma[c] + beta[c]);
    }
  }
  // stage seed_features into fused[:, 0:256]
  for (int i = tid; i < 16 * 256; i += 256) {
    const int mloc = i >> 8, c = i & 255;
    fbuf[mloc][c] = (__bf16)seed_feat[((size_t)b * 256 + c) * MM + m0 + mloc];
  }
  __syncthreads();

  // ---- Stage 2: h = relu(pw1 @ xln + pb1), K=768 ----
  {
    Acc8 a0, a1;
#pragma unroll
    for (int r = 0; r < 8; ++r) { a0.f[r] = 0.f; a1.f[r] = 0.f; }
    const __bf16* wB0 = w1 + (size_t)(w * 16) * DD;
    const __bf16* wB1 = w1 + (size_t)((w + 8) * 16) * DD;
    for (int k0 = 0; k0 < DD; k0 += 32) {
      const v16bf af = load_frag_bf16(&xln[0][0], XLN_LD, k0, lane);
      a0.v = wmma_bf16(af, load_frag_bf16(wB0, DD, k0, lane), a0.v);
      a1.v = wmma_bf16(af, load_frag_bf16(wB1, DD, k0, lane), a1.v);
    }
    {
      const int n0 = w * 16; const float bias = b1[n0 + n];
#pragma unroll
      for (int r = 0; r < 8; ++r)
        hbuf[r + 8 * hi][n0 + n] = (__bf16)fmaxf(a0.f[r] + bias, 0.f);
    }
    {
      const int n0 = (w + 8) * 16; const float bias = b1[n0 + n];
#pragma unroll
      for (int r = 0; r < 8; ++r)
        hbuf[r + 8 * hi][n0 + n] = (__bf16)fmaxf(a1.f[r] + bias, 0.f);
    }
  }
  __syncthreads();

  // ---- Stage 3: vggt_proj = pw2 @ h + pb2 -> fused[:, 256:512], K=256 ----
  {
    Acc8 a0, a1;
#pragma unroll
    for (int r = 0; r < 8; ++r) { a0.f[r] = 0.f; a1.f[r] = 0.f; }
    const __bf16* wB0 = w2 + (size_t)(w * 16) * 256;
    const __bf16* wB1 = w2 + (size_t)((w + 8) * 16) * 256;
    for (int k0 = 0; k0 < 256; k0 += 32) {
      const v16bf af = load_frag_bf16(&hbuf[0][0], H_LD, k0, lane);
      a0.v = wmma_bf16(af, load_frag_bf16(wB0, 256, k0, lane), a0.v);
      a1.v = wmma_bf16(af, load_frag_bf16(wB1, 256, k0, lane), a1.v);
    }
    {
      const int n0 = w * 16; const float bias = b2[n0 + n];
#pragma unroll
      for (int r = 0; r < 8; ++r)
        fbuf[r + 8 * hi][256 + n0 + n] = (__bf16)(a0.f[r] + bias);
    }
    {
      const int n0 = (w + 8) * 16; const float bias = b2[n0 + n];
#pragma unroll
      for (int r = 0; r < 8; ++r)
        fbuf[r + 8 * hi][256 + n0 + n] = (__bf16)(a1.f[r] + bias);
    }
  }
  __syncthreads();

  // ---- Stage 4: h2 = relu(fw1 @ fused + fb1), K=512 -> hbuf ----
  {
    Acc8 a0, a1;
#pragma unroll
    for (int r = 0; r < 8; ++r) { a0.f[r] = 0.f; a1.f[r] = 0.f; }
    const __bf16* wB0 = w3 + (size_t)(w * 16) * 512;
    const __bf16* wB1 = w3 + (size_t)((w + 8) * 16) * 512;
    for (int k0 = 0; k0 < 512; k0 += 32) {
      const v16bf af = load_frag_bf16(&fbuf[0][0], F_LD, k0, lane);
      a0.v = wmma_bf16(af, load_frag_bf16(wB0, 512, k0, lane), a0.v);
      a1.v = wmma_bf16(af, load_frag_bf16(wB1, 512, k0, lane), a1.v);
    }
    __syncthreads();   // hbuf reads in stage 3 are done; safe to overwrite
    {
      const int n0 = w * 16; const float bias = b3[n0 + n];
#pragma unroll
      for (int r = 0; r < 8; ++r)
        hbuf[r + 8 * hi][n0 + n] = (__bf16)fmaxf(a0.f[r] + bias, 0.f);
    }
    {
      const int n0 = (w + 8) * 16; const float bias = b3[n0 + n];
#pragma unroll
      for (int r = 0; r < 8; ++r)
        hbuf[r + 8 * hi][n0 + n] = (__bf16)fmaxf(a1.f[r] + bias, 0.f);
    }
  }
  __syncthreads();

  // ---- Stage 5: mixed = fw2 @ h2 + fb2 -> out [B,256,M] (f32) ----
  {
    Acc8 a0, a1;
#pragma unroll
    for (int r = 0; r < 8; ++r) { a0.f[r] = 0.f; a1.f[r] = 0.f; }
    const __bf16* wB0 = w4 + (size_t)(w * 16) * 256;
    const __bf16* wB1 = w4 + (size_t)((w + 8) * 16) * 256;
    for (int k0 = 0; k0 < 256; k0 += 32) {
      const v16bf af = load_frag_bf16(&hbuf[0][0], H_LD, k0, lane);
      a0.v = wmma_bf16(af, load_frag_bf16(wB0, 256, k0, lane), a0.v);
      a1.v = wmma_bf16(af, load_frag_bf16(wB1, 256, k0, lane), a1.v);
    }
    {
      const int n0 = w * 16; const float bias = b4[n0 + n];
#pragma unroll
      for (int r = 0; r < 8; ++r) {
        const int m = r + 8 * hi;
        out[((size_t)b * 256 + n0 + n) * MM + m0 + m] = a0.f[r] + bias;
      }
    }
    {
      const int n0 = (w + 8) * 16; const float bias = b4[n0 + n];
#pragma unroll
      for (int r = 0; r < 8; ++r) {
        const int m = r + 8 * hi;
        out[((size_t)b * 256 + n0 + n) * MM + m0 + m] = a1.f[r] + bias;
      }
    }
  }
}

// ---------------- host-side launcher ----------------
extern "C" void kernel_launch(void* const* d_in, const int* in_sizes, int n_in,
                              void* d_out, int out_size, void* d_ws, size_t ws_size,
                              hipStream_t stream) {
  const float* pc         = (const float*)d_in[0];
  const float* seed_xyz   = (const float*)d_in[1];
  const float* seed_feats = (const float*)d_in[2];
  const float* pts        = (const float*)d_in[3];
  const float* feat       = (const float*)d_in[4];
  const float* ln_gamma   = (const float*)d_in[5];
  const float* ln_beta    = (const float*)d_in[6];
  const float* pw1        = (const float*)d_in[7];
  const float* pb1        = (const float*)d_in[8];
  const float* pw2        = (const float*)d_in[9];
  const float* pb2        = (const float*)d_in[10];
  const float* fw1        = (const float*)d_in[11];
  const float* fb1        = (const float*)d_in[12];
  const float* fw2        = (const float*)d_in[13];
  const float* fb2        = (const float*)d_in[14];

  char* ws = (char*)d_ws;
  float*  cs     = (float*)(ws + 0);          //  64 B
  float*  seed_n = (float*)(ws + 256);        //  49152 B
  float*  pts_n  = (float*)(ws + 49408);      // 786432 B
  int*    idx    = (int*)  (ws + 835840);     //  16384 B
  __bf16* w1bf   = (__bf16*)(ws + 852224);    // 393216 B (256x768)
  __bf16* w2bf   = (__bf16*)(ws + 1245440);   // 131072 B (256x256)
  __bf16* w3bf   = (__bf16*)(ws + 1376512);   // 262144 B (256x512)
  __bf16* w4bf   = (__bf16*)(ws + 1638656);   // 131072 B (256x256)

  centroid_kernel<<<BB, 256, 0, stream>>>(pc, cs);

  const int ntot = BB * (KK + MM);
  normalize_kernel<<<(ntot + 255) / 256, 256, 0, stream>>>(pts, seed_xyz, cs,
                                                           pts_n, seed_n);

  nn_kernel<<<BB * MM / 8, 256, 0, stream>>>(seed_n, pts_n, idx);

  cvt_bf16_kernel<<<(256 * 768 + 255) / 256, 256, 0, stream>>>(pw1, w1bf, 256 * 768);
  cvt_bf16_kernel<<<(256 * 256 + 255) / 256, 256, 0, stream>>>(pw2, w2bf, 256 * 256);
  cvt_bf16_kernel<<<(256 * 512 + 255) / 256, 256, 0, stream>>>(fw1, w3bf, 256 * 512);
  cvt_bf16_kernel<<<(256 * 256 + 255) / 256, 256, 0, stream>>>(fw2, w4bf, 256 * 256);

  fused_mlp_kernel<<<BB * (MM / 16), 256, 0, stream>>>(
      feat, idx, ln_gamma, ln_beta,
      w1bf, pb1, w2bf, pb2, w3bf, fb1, w4bf, fb2,
      seed_feats, (float*)d_out);
}
